// Convolution_22917945491528
// MI455X (gfx1250) — matrix operand order repaired and checked by reading, not compiled, not run on hardware
//
#include <hip/hip_runtime.h>
#include <math.h>

typedef __attribute__((ext_vector_type(16))) _Float16 v16h;
typedef __attribute__((ext_vector_type(8)))  _Float16 v8h;
typedef __attribute__((ext_vector_type(8)))  float    v8f;
typedef __attribute__((ext_vector_type(4)))  unsigned int u32x4;
typedef __attribute__((ext_vector_type(8)))  unsigned int u32x8;

#define N_NODES 10000
#define N_EDGES 160000

__device__ __forceinline__ void atomAdd(float* p, float v) {
  unsafeAtomicAdd(p, v);   // hardware global_atomic_add_f32, no return
}

// ---------------------------------------------------------------------------
// Kernel 0: one-time fc_w1 (64x320 f32) -> w1T (320x64 f16, transposed)
// ---------------------------------------------------------------------------
__global__ __launch_bounds__(256) void w1_prep_kernel(
    const float* __restrict__ fc_w1, _Float16* __restrict__ w1T)
{
  const int i = blockIdx.x*256 + threadIdx.x;   // 20480 total
  const int n = i >> 6, k = i & 63;
  w1T[i] = (_Float16)fc_w1[k*320 + n];
}

// ---------------------------------------------------------------------------
// Kernel 1: per-node self-connection s (-> out, scaled by sin(pi/8)) and
// y = lin1(x) (-> workspace), weights staged in LDS.
// ---------------------------------------------------------------------------
__global__ __launch_bounds__(256) void node_prep_kernel(
    const float* __restrict__ node_input, const float* __restrict__ node_attr,
    const float* __restrict__ sc_w0, const float* __restrict__ sc_w1,
    const float* __restrict__ lin1_w0, const float* __restrict__ lin1_w1,
    float* __restrict__ out, float* __restrict__ y)
{
  __shared__ float s_scw0[64*64];
  __shared__ float s_scw1[32*32];
  __shared__ float s_l1w0[64*64];
  __shared__ float s_l1w1[32*32];
  for (int i = threadIdx.x; i < 64*64; i += 256) { s_scw0[i]=sc_w0[i]; s_l1w0[i]=lin1_w0[i]; }
  for (int i = threadIdx.x; i < 32*32; i += 256) { s_scw1[i]=sc_w1[i]; s_l1w1[i]=lin1_w1[i]; }
  __syncthreads();
  const int n = blockIdx.x*256 + threadIdx.x;
  if (n >= N_NODES) return;
  const float a  = node_attr[n];
  const float* xi = node_input + (size_t)n*160;
  float* on = out + (size_t)n*320;
  float* yn = y   + (size_t)n*160;
  const float cs = 0.38268343f;             // sin(pi/8)
  float x0[64];
  #pragma unroll
  for (int u=0;u<64;u++) x0[u]=xi[u];
  const float f0 = a * 0.125f;              // 1/sqrt(64)
  for (int v=0;v<64;v++){
    float as=0.f, ay=0.f;
    #pragma unroll
    for (int u=0;u<64;u++){ as += x0[u]*s_scw0[u*64+v]; ay += x0[u]*s_l1w0[u*64+v]; }
    on[v] = cs*as*f0;
    yn[v] = ay*f0;
  }
  float x1[96];
  #pragma unroll
  for (int u=0;u<96;u++) x1[u]=xi[64+u];
  const float f1 = a * 0.17677670f;         // 1/sqrt(32)
  for (int v=0;v<32;v++){
    float as0=0,as1=0,as2=0, ay0=0,ay1=0,ay2=0;
    #pragma unroll
    for (int u=0;u<32;u++){
      const float ws_=s_scw1[u*32+v], wy=s_l1w1[u*32+v];
      as0+=x1[u*3+0]*ws_; as1+=x1[u*3+1]*ws_; as2+=x1[u*3+2]*ws_;
      ay0+=x1[u*3+0]*wy;  ay1+=x1[u*3+1]*wy;  ay2+=x1[u*3+2]*wy;
    }
    on[64+v*3+0]=cs*as0*f1; on[64+v*3+1]=cs*as1*f1; on[64+v*3+2]=cs*as2*f1;
    yn[64+v*3+0]=ay0*f1;    yn[64+v*3+1]=ay1*f1;    yn[64+v*3+2]=ay2*f1;
  }
  for (int c=160;c<320;c++) on[c]=0.f;      // l=2 part of s is zero
}

// ---------------------------------------------------------------------------
// Kernel 2: fused edge MLP (WMMA f16->f32) + tensor product + atomic scatter.
// 128 threads = 4 waves, 16 edges per wave, 64 edges per block.
// B-matrix (40KB f16) is DMA'd into LDS by the Tensor Data Mover.
// ---------------------------------------------------------------------------
__global__ __launch_bounds__(128) void edge_kernel(
    const int* __restrict__ edge_src, const int* __restrict__ edge_dst,
    const float* __restrict__ edge_attr, const float* __restrict__ ele,
    const float* __restrict__ fc_w0, const _Float16* __restrict__ w1T_g,
    const float* __restrict__ y, float* __restrict__ agg)
{
  __shared__ float    s_fcw0[8*64];      // 2 KB
  __shared__ _Float16 s_w1T[320*64];     // 40 KB, fc_w1 transposed [n][k], f16
  __shared__ _Float16 s_h[64*64];        // 8 KB, h activations, f16
  __shared__ int      s_src[64], s_dst[64];
  __shared__ float    s_ea[64*9];

  const int tid   = threadIdx.x;
  const int eBase = blockIdx.x * 64;
  const int wv    = tid >> 5;

  // --- TDM: each wave DMAs its 10KB quarter of w1T from global into LDS ---
  {
    const unsigned wvu = (unsigned)__builtin_amdgcn_readfirstlane(wv);
    const unsigned lds_base = (unsigned)(size_t)(&s_w1T[0]) + wvu*10240u;
    const unsigned long long ga =
        (unsigned long long)(size_t)(w1T_g + (size_t)wvu*5120u);
    u32x4 g0;
    g0[0] = 1u;                                          // count=1, user desc
    g0[1] = lds_base;                                    // lds_addr (bytes)
    g0[2] = (unsigned)(ga & 0xFFFFFFFFu);                // global_addr[31:0]
    g0[3] = (unsigned)((ga >> 32) & 0x01FFFFFFu)         // global_addr[56:32]
          | 0x80000000u;                                 // type=2 ("image")
    u32x8 g1;
    g1[0] = 0x00010000u;                                 // data_size=1 (2 bytes)
    g1[1] = (5120u & 0xFFFFu) << 16;                     // tensor_dim0[15:0]
    g1[2] = 1u << 16;                                    // tensor_dim1 = 1
    g1[3] = (5120u & 0xFFFFu) << 16;                     // tile_dim0 = 5120
    g1[4] = 1u;                                          // tile_dim1 = 1
    g1[5] = 5120u;                                       // tensor_dim0_stride
    g1[6] = 0u;
    g1[7] = 0u;
    asm volatile("tensor_load_to_lds %0, %1" :: "s"(g0), "s"(g1) : "memory");
    __builtin_amdgcn_s_wait_tensorcnt(0);
  }

  for (int i=tid;i<512;i+=128) s_fcw0[i]=fc_w0[i];
  if (tid < 64){
    const int e = eBase+tid;
    s_src[tid]=edge_src[e]; s_dst[tid]=edge_dst[e];
    #pragma unroll
    for(int j=0;j<9;j++) s_ea[tid*9+j]=edge_attr[(size_t)e*9+j];
  }
  __syncthreads();

  // h = silu(ele @ fc_w0 / sqrt(8)) * SILU_C   (2 threads per edge, 32 cols each)
  {
    const int el = tid>>1;
    const int cb = (tid&1)*32;
    const float* ep = ele + (size_t)(eBase+el)*8;
    float e8[8];
    #pragma unroll
    for(int j=0;j<8;j++) e8[j]=ep[j];
    for(int c0=0;c0<32;c0++){
      const int c=cb+c0;
      float acc=0.f;
      #pragma unroll
      for(int j=0;j<8;j++) acc += e8[j]*s_fcw0[j*64+c];
      acc *= 0.35355339f;                           // 1/sqrt(8)
      const float hv = acc/(1.f+__expf(-acc))*1.6791f;  // silu * SILU_C
      s_h[el*64+c]=(_Float16)hv;
    }
  }
  __syncthreads();

  const int lane  = tid & 31;
  const int n15   = lane & 15;
  const int khalf = lane >> 4;
  const int le0   = wv*16;

  // A fragments: contiguous k order (matches B); lane half h holds K = 16h+0..15
  v16h a0, a1;
  {
    const _Float16* hp = &s_h[(le0 + n15)*64];
    const v8h A0l = *(const v8h*)(hp + khalf*16);
    const v8h A0h = *(const v8h*)(hp + 8 + khalf*16);
    const v8h A1l = *(const v8h*)(hp + 32 + khalf*16);
    const v8h A1h = *(const v8h*)(hp + 40 + khalf*16);
    #pragma unroll
    for(int i=0;i<8;i++){ a0[i]=A0l[i]; a0[i+8]=A0h[i]; a1[i]=A1l[i]; a1[i+8]=A1h[i]; }
  }

  const float WSCALE = 0.03125f;  // 1/sqrt(64) (fc) * 1/sqrt(16) (avg degree)
  for (int t=0;t<20;t++){
    // B fragment: col = t*16 + n15, K = khalf*16 + 0..15 (+32 for step 1)
    const _Float16* bp = &s_w1T[(t*16+n15)*64 + khalf*16];
    const v16h b0 = *(const v16h*)bp;
    const v16h b1 = *(const v16h*)(bp+32);
    v8f c = {};
    c = __builtin_amdgcn_wmma_f32_16x16x32_f16(false,a0,false,b0,(short)0,c,false,false);
    c = __builtin_amdgcn_wmma_f32_16x16x32_f16(false,a1,false,b1,(short)0,c,false,false);
    const int slot = t*16+n15;
    #pragma unroll
    for(int r=0;r<8;r++){
      const int le  = le0 + r + 8*khalf;      // D-matrix row M = r + 8*half
      const int src = s_src[le];
      const int dst = s_dst[le];
      const float* ea = &s_ea[le*9];
      float* arow = agg + (size_t)dst*960;
      const float* yrow = y + (size_t)src*160;
      const float wvv = c[r]*WSCALE;
      if (t<4){                               // k0: 0e x 0o -> 0e  (W000=1)
        const int u=slot;
        atomAdd(&arow[u], wvv*yrow[u]*ea[0]);
      } else if (t<8){                        // k2: 0e x 1o -> 1o  (sqrt3*delta/sqrt3)
        const int u=slot-64;
        const float f=wvv*yrow[u];
        atomAdd(&arow[96+u*3+0], f*ea[1]);
        atomAdd(&arow[96+u*3+1], f*ea[2]);
        atomAdd(&arow[96+u*3+2], f*ea[3]);
      } else if (t<12){                       // k5: 0e x 2e -> 2e
        const int u=slot-128;
        const float f=wvv*yrow[u];
        #pragma unroll
        for(int k=0;k<5;k++) atomAdd(&arow[480+u*5+k], f*ea[4+k]);
      } else if (t<14){                       // k3: 1o x 0o -> 1o
        const int u=slot-192;
        const float f=wvv*ea[0];
        #pragma unroll
        for(int k=0;k<3;k++) atomAdd(&arow[288+u*3+k], f*yrow[64+u*3+k]);
      } else if (t<16){                       // k1: 1o . 1o -> 0e  (delta/sqrt3)
        const int u=slot-224;
        const float d = yrow[64+u*3]*ea[1]+yrow[64+u*3+1]*ea[2]+yrow[64+u*3+2]*ea[3];
        atomAdd(&arow[64+u], wvv*0.57735027f*d);
      } else if (t<18){                       // k6: 1o x 1o -> 2e  (sqrt5 * W112)
        const int u=slot-256;
        const float v0=yrow[64+u*3], v1=yrow[64+u*3+1], v2=yrow[64+u*3+2];
        const float b0_=ea[1], b1_=ea[2], b2_=ea[3];
        const float A10=0.70710678f, A30=0.40824829f; // sqrt5/sqrt10, sqrt5/sqrt30
        atomAdd(&arow[800+u*5+0], wvv*(-A10)*(v0*b2_+v2*b0_));
        atomAdd(&arow[800+u*5+1], wvv*(-A10)*(v0*b1_+v1*b0_));
        atomAdd(&arow[800+u*5+2], wvv*  A30 *(v0*b0_-2.f*v1*b1_+v2*b2_));
        atomAdd(&arow[800+u*5+3], wvv*(-A10)*(v1*b2_+v2*b1_));
        atomAdd(&arow[800+u*5+4], wvv*  A10 *(v0*b0_-v2*b2_));
      } else {                                // k4: 1o x 2e -> 1o  (sqrt3 * W121)
        const int u=slot-288;
        const float v0=yrow[64+u*3], v1=yrow[64+u*3+1], v2=yrow[64+u*3+2];
        const float g0=ea[4],g1=ea[5],g2=ea[6],g3=ea[7],g4=ea[8];
        const float B10=0.54772256f, B30=0.31622777f; // sqrt3/sqrt10, sqrt3/sqrt30
        atomAdd(&arow[384+u*3+0], wvv*(-B10*(v2*g0+v1*g1)+B30*v0*g2+B10*v0*g4));
        atomAdd(&arow[384+u*3+1], wvv*(-B10*(v0*g1+v2*g3)-2.f*B30*v1*g2));
        atomAdd(&arow[384+u*3+2], wvv*(-B10*(v0*g0+v1*g3)+B30*v2*g2-B10*v2*g4));
      }
    }
  }
}

// ---------------------------------------------------------------------------
// Kernel 3: lin2 on aggregated messages, masked add into out.
// One block = 8 nodes, one thread = one of 320 output channels.
// ---------------------------------------------------------------------------
__global__ __launch_bounds__(320) void node_out_kernel(
    const float* __restrict__ agg, const float* __restrict__ node_attr,
    const float* __restrict__ lin2_w0, const float* __restrict__ lin2_w1,
    const float* __restrict__ lin2_w2, float* __restrict__ out)
{
  __shared__ float s_w0[96*64];
  __shared__ float s_w1[128*32];
  __shared__ float s_w2[96*32];
  const int tid = threadIdx.x;
  for(int i=tid;i<96*64;i+=320)  s_w0[i]=lin2_w0[i];
  for(int i=tid;i<128*32;i+=320) s_w1[i]=lin2_w1[i];
  for(int i=tid;i<96*32;i+=320)  s_w2[i]=lin2_w2[i];
  __syncthreads();
  const float cx=0.92387953f;    // cos(pi/8)
  const int c = tid;
  for (int nn=0;nn<8;nn++){
    const int n = blockIdx.x*8+nn;
    const float a = node_attr[n];
    const float* ar = agg + (size_t)n*960;
    float val, fac;
    if (c<64){
      float acc=0.f;
      for(int u=0;u<96;u++) acc += ar[u]*s_w0[u*64+c];
      val = acc*0.10206207f; fac = cx;               // /sqrt(96)
    } else if (c<160){
      const int q=c-64, v=q/3, i=q%3;
      float acc=0.f;
      for(int u=0;u<128;u++) acc += ar[96+u*3+i]*s_w1[u*32+v];
      val = acc*0.08838835f; fac = cx;               // /sqrt(128)
    } else {
      const int q=c-160, v=q/5, i=q%5;
      float acc=0.f;
      for(int u=0;u<96;u++) acc += ar[480+u*5+i]*s_w2[u*32+v];
      val = acc*0.10206207f; fac = 1.0f;             // l=2: mask passthrough
    }
    out[(size_t)n*320+c] += fac*a*val;
  }
}

// ---------------------------------------------------------------------------
extern "C" void kernel_launch(void* const* d_in, const int* in_sizes, int n_in,
                              void* d_out, int out_size, void* d_ws, size_t ws_size,
                              hipStream_t stream)
{
  const float* node_input = (const float*)d_in[0];
  const float* node_attr  = (const float*)d_in[1];
  const int*   edge_src   = (const int*)d_in[2];
  const int*   edge_dst   = (const int*)d_in[3];
  const float* edge_attr  = (const float*)d_in[4];
  const float* ele        = (const float*)d_in[5];
  const float* sc_w0      = (const float*)d_in[6];
  const float* sc_w1      = (const float*)d_in[7];
  const float* lin1_w0    = (const float*)d_in[8];
  const float* lin1_w1    = (const float*)d_in[9];
  const float* fc_w0      = (const float*)d_in[10];
  const float* fc_w1      = (const float*)d_in[11];
  const float* lin2_w0    = (const float*)d_in[12];
  const float* lin2_w1    = (const float*)d_in[13];
  const float* lin2_w2    = (const float*)d_in[14];
  float* out = (float*)d_out;

  float*    y    = (float*)d_ws;                                     // N x 160 f32
  float*    agg  = (float*)((char*)d_ws + (size_t)N_NODES*160*4);    // N x 960 f32
  _Float16* w1T  = (_Float16*)((char*)d_ws + (size_t)N_NODES*160*4
                                           + (size_t)N_NODES*960*4); // 320x64 f16

  hipMemsetAsync(agg, 0, (size_t)N_NODES*960*sizeof(float), stream);
  w1_prep_kernel<<<20480/256, 256, 0, stream>>>(fc_w1, w1T);
  node_prep_kernel<<<(N_NODES+255)/256, 256, 0, stream>>>(
      node_input, node_attr, sc_w0, sc_w1, lin1_w0, lin1_w1, out, y);
  edge_kernel<<<N_EDGES/64, 128, 0, stream>>>(
      edge_src, edge_dst, edge_attr, ele, fc_w0, w1T, y, agg);
  node_out_kernel<<<N_NODES/8, 320, 0, stream>>>(
      agg, node_attr, lin2_w0, lin2_w1, lin2_w2, out);
}
